// RecurrentPolicy_15092515078167
// MI455X (gfx1250) — matrix operand order repaired and checked by reading, not compiled, not run on hardware
//
#include <hip/hip_runtime.h>
#include <hip/hip_bf16.h>

// ---------------------------------------------------------------------------
// RecurrentPolicy forward on gfx1250 (MI455X), all GEMMs via
// v_wmma_f32_16x16x32_bf16 (wave32 WMMA, f32 accumulate).
//
// v2 changes:
//  * GRU step writes PRE-MASKED h (bf16 + f32) using done[t], so the next
//    step's A-fragments are direct v8bf loads: no f32 loads, no cvt, no mask
//    work on the serial critical path.
//  * Gate nonlinearities use raw v_exp_f32 / v_rcp_f32 (branch-free) instead
//    of libm expf/tanhf.
//  * global_prefetch of the next step's gi tile.
// ---------------------------------------------------------------------------

typedef __attribute__((ext_vector_type(16))) __bf16 v16bf;
typedef __attribute__((ext_vector_type(8)))  __bf16 v8bf;
typedef __attribute__((ext_vector_type(8)))  float  v8f;

#define T_STEPS 512
#define BATCH   256
#define OBSD    128
#define HID     256
#define ACT     16
#define G3      768   // 3*HID

#define LOG2E  1.4426950408889634f

__device__ __forceinline__ __bf16 f2bf(float f) { return (__bf16)f; }

// branch-free hardware transcendentals
__device__ __forceinline__ float sigm(float x) {
  return __builtin_amdgcn_rcpf(1.0f + __builtin_amdgcn_exp2f(-LOG2E * x));
}
__device__ __forceinline__ float fast_tanh(float x) {
  return 2.0f * __builtin_amdgcn_rcpf(1.0f + __builtin_amdgcn_exp2f(-2.0f * LOG2E * x)) - 1.0f;
}

__device__ __forceinline__ v8f wmma_bf16(v16bf a, v16bf b, v8f c) {
  // (neg_a, A, neg_b, B, c_mod, C, reuse_a, reuse_b)
  return __builtin_amdgcn_wmma_f32_16x16x32_bf16(false, a, false, b, (short)0, c,
                                                 false, false);
}

// build a 16x32 bf16 A-fragment from a row of contiguous bf16 memory
__device__ __forceinline__ v16bf load_afrag_bf(const __bf16* row, int k0) {
  v8bf lo = *(const v8bf*)(row + k0);
  v8bf hh = *(const v8bf*)(row + k0 + 8);
  v16bf a;
  #pragma unroll
  for (int i = 0; i < 8; ++i) { a[i] = lo[i]; a[8 + i] = hh[i]; }
  return a;
}

// ---------------------------------------------------------------------------
// Kernel 0: convert weights f32 -> bf16 (row-major [out,in] == B-matrix
// column-major), build padded head weights, zero masked-h state buffers.
// Re-run every call => deterministic across graph replays.
// ---------------------------------------------------------------------------
__global__ void __launch_bounds__(256)
prep_kernel(const float* __restrict__ encW, const float* __restrict__ Wih,
            const float* __restrict__ Whh, const float* __restrict__ polW,
            const float* __restrict__ valW,
            __bf16* __restrict__ encWb, __bf16* __restrict__ Wihb,
            __bf16* __restrict__ Whhb, __bf16* __restrict__ headWb,
            __bf16* __restrict__ hmb0, float* __restrict__ hmf0) {
  int i = blockIdx.x * blockDim.x + threadIdx.x;
  if (i < HID * OBSD) encWb[i] = f2bf(encW[i]);
  if (i < G3 * HID) { Wihb[i] = f2bf(Wih[i]); Whhb[i] = f2bf(Whh[i]); }
  if (i < 32 * HID) {           // rows 0..15 = pol_W, row 16 = val_W, rest 0
    int r = i >> 8, c = i & 255;
    float v = (r < ACT) ? polW[r * HID + c] : (r == ACT ? valW[c] : 0.0f);
    headWb[i] = f2bf(v);
  }
  if (i < BATCH * HID) { hmf0[i] = 0.0f; hmb0[i] = f2bf(0.0f); }
}

// ---------------------------------------------------------------------------
// Kernel 1: fused encoder + GRU input projection.
//   x  = tanh(obs @ enc_W^T + enc_b)        [16-row tile, K=128]
//   gi = x @ W_ih^T + b_ih                  [K=256] -> global f32
// One workgroup (8 waves) per 16-row tile of T*B. x tile lives in LDS (bf16).
// ---------------------------------------------------------------------------
__global__ void __launch_bounds__(256)
enc_gi_kernel(const float* __restrict__ obs, const float* __restrict__ enc_b,
              const __bf16* __restrict__ encWb, const __bf16* __restrict__ Wihb,
              const float* __restrict__ b_ih, float* __restrict__ gi) {
  __shared__ __align__(16) __bf16 xlds[16][HID + 8];  // +16B pad per row
  const int lane = threadIdx.x & 31;
  const int wave = threadIdx.x >> 5;
  const int col  = lane & 15;
  const int hi   = lane >> 4;            // 0 or 1 (half-wave)
  const size_t mbase = (size_t)blockIdx.x * 16;

  // ---- stage 1: A fragments of obs (K=128 -> 4 frags), reused by both ntiles
  v16bf afr[4];
  {
    const float* ap = obs + (mbase + col) * OBSD;
    #pragma unroll
    for (int kk = 0; kk < 4; ++kk) {
      const int k0 = kk * 32 + hi * 16;
      float4 a0 = *(const float4*)(ap + k0);
      float4 a1 = *(const float4*)(ap + k0 + 4);
      float4 a2 = *(const float4*)(ap + k0 + 8);
      float4 a3 = *(const float4*)(ap + k0 + 12);
      v16bf a;
      a[0]  = f2bf(a0.x); a[1]  = f2bf(a0.y); a[2]  = f2bf(a0.z); a[3]  = f2bf(a0.w);
      a[4]  = f2bf(a1.x); a[5]  = f2bf(a1.y); a[6]  = f2bf(a1.z); a[7]  = f2bf(a1.w);
      a[8]  = f2bf(a2.x); a[9]  = f2bf(a2.y); a[10] = f2bf(a2.z); a[11] = f2bf(a2.w);
      a[12] = f2bf(a3.x); a[13] = f2bf(a3.y); a[14] = f2bf(a3.z); a[15] = f2bf(a3.w);
      afr[kk] = a;
    }
  }
  #pragma unroll
  for (int s = 0; s < 2; ++s) {          // 16 x-tiles over 8 waves
    const int ntile = wave * 2 + s;
    v8f acc = {};
    #pragma unroll
    for (int kk = 0; kk < 4; ++kk) {
      const __bf16* bp = encWb + (size_t)(ntile * 16 + col) * OBSD + kk * 32 + hi * 16;
      acc = wmma_bf16(afr[kk], *(const v16bf*)bp, acc);
    }
    const float bias = enc_b[ntile * 16 + col];
    #pragma unroll
    for (int i = 0; i < 8; ++i)
      xlds[hi * 8 + i][ntile * 16 + col] = f2bf(fast_tanh(acc[i] + bias));
  }
  __syncthreads();

  // ---- stage 2: gi (48 n-tiles over 8 waves, K=256 -> 8 frags from LDS)
  v16bf xfr[8];
  #pragma unroll
  for (int kk = 0; kk < 8; ++kk)
    xfr[kk] = load_afrag_bf(&xlds[col][0], kk * 32 + hi * 16);

  #pragma unroll
  for (int s = 0; s < 6; ++s) {
    const int ntile = wave * 6 + s;
    v8f acc = {};
    #pragma unroll
    for (int kk = 0; kk < 8; ++kk) {
      const __bf16* bp = Wihb + (size_t)(ntile * 16 + col) * HID + kk * 32 + hi * 16;
      acc = wmma_bf16(xfr[kk], *(const v16bf*)bp, acc);
    }
    const float bias = b_ih[ntile * 16 + col];
    #pragma unroll
    for (int i = 0; i < 8; ++i)
      gi[(mbase + hi * 8 + i) * G3 + ntile * 16 + col] = acc[i] + bias;
  }
}

// ---------------------------------------------------------------------------
// Kernel 2: one GRU step. One wave per 16x16 h_new tile (256 waves total).
// Inputs are PRE-MASKED hidden state from step t-1:
//   hmb_in (bf16, A-matrix ready) / hmf_in (f32)  = h_{t-1} * (1 - done[t-1])
//   gh = hm @ W_hh^T + b_hh ; gates ; h_new = (1-z)*n + z*hm
// Outputs masked state for step t+1 using done[t], plus unmasked outs (bf16).
// ---------------------------------------------------------------------------
__global__ void __launch_bounds__(128)
gru_step_kernel(int t, const float* __restrict__ done, const float* __restrict__ gi,
                const __bf16* __restrict__ Whhb, const float* __restrict__ b_hh,
                const __bf16* __restrict__ hmb_in, const float* __restrict__ hmf_in,
                __bf16* __restrict__ hmb_out, float* __restrict__ hmf_out,
                __bf16* __restrict__ outs) {
  const int lane  = threadIdx.x & 31;
  const int wave  = threadIdx.x >> 5;
  const int tile  = blockIdx.x * 4 + wave;   // 0..255
  const int btile = tile >> 4;
  const int ntile = tile & 15;
  const int col   = lane & 15;
  const int hi    = lane >> 4;
  const int h     = ntile * 16 + col;        // hidden column of this lane

  const __bf16* ap = hmb_in + (size_t)(btile * 16 + col) * HID;  // A row

  v8f accR = {}, accZ = {}, accN = {};
  #pragma unroll
  for (int kk = 0; kk < 8; ++kk) {
    v16bf a = load_afrag_bf(ap, kk * 32 + hi * 16);
    const int kb = kk * 32 + hi * 16;
    const __bf16* bR = Whhb + (size_t)(0 * HID + h) * HID + kb;
    const __bf16* bZ = Whhb + (size_t)(1 * HID + h) * HID + kb;
    const __bf16* bN = Whhb + (size_t)(2 * HID + h) * HID + kb;
    accR = wmma_bf16(a, *(const v16bf*)bR, accR);
    accZ = wmma_bf16(a, *(const v16bf*)bZ, accZ);
    accN = wmma_bf16(a, *(const v16bf*)bN, accN);
  }
  const float bhr = b_hh[h], bhz = b_hh[HID + h], bhn = b_hh[2 * HID + h];
  #pragma unroll
  for (int i = 0; i < 8; ++i) {
    const int b = btile * 16 + hi * 8 + i;   // batch row of c[i]
    const size_t gib = ((size_t)t * BATCH + b) * G3;
    const float r  = sigm(gi[gib + h]            + accR[i] + bhr);
    const float z  = sigm(gi[gib + HID + h]      + accZ[i] + bhz);
    const float n  = fast_tanh(gi[gib + 2 * HID + h] + r * (accN[i] + bhn));
    const float hm = hmf_in[(size_t)b * HID + h];        // already masked
    const float hn = (1.0f - z) * n + z * hm;
    const float md = 1.0f - done[(size_t)t * BATCH + b]; // mask for step t+1
    hmf_out[(size_t)b * HID + h] = hn * md;
    hmb_out[(size_t)b * HID + h] = f2bf(hn * md);
    outs[((size_t)t * BATCH + b) * HID + h] = f2bf(hn);
  }
  // keep next step's gate inputs L2-hot (global_prefetch_b8)
  if (t + 1 < T_STEPS)
    __builtin_prefetch(gi + ((size_t)(t + 1) * BATCH + btile * 16 + hi * 8) * G3 + h, 0, 1);
}

// ---------------------------------------------------------------------------
// Kernel 3: policy/value heads.  headWb = [pol_W(16) ; val_W(1) ; zeros] as
// a 32x256 bf16 block -> 2 n-tiles per 16-row m-tile, K=256.
// ---------------------------------------------------------------------------
__global__ void __launch_bounds__(128)
head_kernel(const __bf16* __restrict__ outs, const __bf16* __restrict__ headWb,
            const float* __restrict__ pol_b, const float* __restrict__ val_b,
            float* __restrict__ out) {
  const int lane  = threadIdx.x & 31;
  const int wave  = threadIdx.x >> 5;
  const int w     = blockIdx.x * 4 + wave;   // 0..16383
  const int mtile = w >> 1;
  const int nsel  = w & 1;
  const int col   = lane & 15;
  const int hi    = lane >> 4;
  const size_t mbase = (size_t)mtile * 16;

  const __bf16* ap = outs + (mbase + col) * HID;
  v8f acc = {};
  #pragma unroll
  for (int kk = 0; kk < 8; ++kk) {
    v16bf a = load_afrag_bf(ap, kk * 32 + hi * 16);
    const __bf16* bp = headWb + (size_t)(nsel * 16 + col) * HID + kk * 32 + hi * 16;
    acc = wmma_bf16(a, *(const v16bf*)bp, acc);
  }
  if (nsel == 0) {
    const float bias = pol_b[col];
    #pragma unroll
    for (int i = 0; i < 8; ++i)
      out[(mbase + hi * 8 + i) * ACT + col] = acc[i] + bias;
  } else if (col == 0) {
    const float bias = val_b[0];
    #pragma unroll
    for (int i = 0; i < 8; ++i)
      out[(size_t)T_STEPS * BATCH * ACT + mbase + hi * 8 + i] = acc[i] + bias;
  }
}

// ---------------------------------------------------------------------------
extern "C" void kernel_launch(void* const* d_in, const int* in_sizes, int n_in,
                              void* d_out, int out_size, void* d_ws, size_t ws_size,
                              hipStream_t stream) {
  (void)in_sizes; (void)n_in; (void)out_size; (void)ws_size;
  const float* obs   = (const float*)d_in[0];
  const float* done  = (const float*)d_in[1];
  const float* encW  = (const float*)d_in[2];
  const float* enc_b = (const float*)d_in[3];
  const float* Wih   = (const float*)d_in[4];
  const float* Whh   = (const float*)d_in[5];
  const float* b_ih  = (const float*)d_in[6];
  const float* b_hh  = (const float*)d_in[7];
  const float* polW  = (const float*)d_in[8];
  const float* pol_b = (const float*)d_in[9];
  const float* valW  = (const float*)d_in[10];
  const float* val_b = (const float*)d_in[11];
  float* out = (float*)d_out;

  // workspace carve-up (256B aligned)
  char* p = (char*)d_ws;
  auto carve = [&](size_t bytes) -> char* {
    char* r = p; p += (bytes + 255) & ~(size_t)255; return r;
  };
  __bf16* encWb  = (__bf16*)carve((size_t)HID * OBSD * 2);
  __bf16* Wihb   = (__bf16*)carve((size_t)G3 * HID * 2);
  __bf16* Whhb   = (__bf16*)carve((size_t)G3 * HID * 2);
  __bf16* headWb = (__bf16*)carve((size_t)32 * HID * 2);
  __bf16* hmb0   = (__bf16*)carve((size_t)BATCH * HID * 2);   // masked h, bf16
  __bf16* hmb1   = (__bf16*)carve((size_t)BATCH * HID * 2);
  float*  hmf0   = (float*)carve((size_t)BATCH * HID * 4);    // masked h, f32
  float*  hmf1   = (float*)carve((size_t)BATCH * HID * 4);
  float*  gi     = (float*)carve((size_t)T_STEPS * BATCH * G3 * 4);
  __bf16* outsb  = (__bf16*)carve((size_t)T_STEPS * BATCH * HID * 2);

  // 0) weight conversion + initial (zero) masked-state buffers
  prep_kernel<<<(G3 * HID + 255) / 256, 256, 0, stream>>>(
      encW, Wih, Whh, polW, valW, encWb, Wihb, Whhb, headWb, hmb0, hmf0);

  // 1) time-parallel encoder + input projection
  enc_gi_kernel<<<(T_STEPS * BATCH) / 16, 256, 0, stream>>>(
      obs, enc_b, encWb, Wihb, b_ih, gi);

  // 2) sequential GRU scan: 512 small step kernels, ping-pong masked state
  __bf16* hmb[2] = {hmb0, hmb1};
  float*  hmf[2] = {hmf0, hmf1};
  for (int t = 0; t < T_STEPS; ++t) {
    gru_step_kernel<<<64, 128, 0, stream>>>(
        t, done, gi, Whhb, b_hh,
        hmb[t & 1], hmf[t & 1], hmb[(t + 1) & 1], hmf[(t + 1) & 1], outsb);
  }

  // 3) heads
  head_kernel<<<((T_STEPS * BATCH / 16) * 2) / 4, 128, 0, stream>>>(
      outsb, headWb, pol_b, val_b, out);
}
